// SpatialGNN_67705864454403
// MI455X (gfx1250) — matrix operand order repaired and checked by reading, not compiled, not run on hardware
//
#include <hip/hip_runtime.h>
#include <math.h>

// ---------------------------------------------------------------------------
// SpatialGNN for MI455X (gfx1250): 3x GCN + GAT(4 heads) + MLP + log_softmax
// Dense GEMMs (K in {64,256}) via v_wmma_f32_16x16x32_f16, wave32,
// 1 wave = one 16x16 output tile, B pre-packed to the WMMA lane layout.
// Edge scatter/gather via f32 global atomics (the memory-bound part).
// ---------------------------------------------------------------------------

typedef __attribute__((ext_vector_type(16))) _Float16 v16h;
typedef __attribute__((ext_vector_type(8)))  float    v8f;

#define HID   64
#define HEADS 4
#define GDIM  256   // HEADS*HID
#define OUTD  3
#define TPB   256

static inline int cdiv_h(long a, long b) { return (int)((a + b - 1) / b); }

__device__ __forceinline__ float lrelu02(float x) { return x > 0.f ? x : 0.2f * x; }

// float atomic max via ordered integer encoding (location always pre-seeded
// with a real value, so mixed-sign two-op scheme is exact).
__device__ __forceinline__ void atomicMaxF(float* addr, float v) {
    if (v >= 0.f) atomicMax((int*)addr, __float_as_int(v));
    else          atomicMin((unsigned int*)addr, __float_as_uint(v));
}

// ---------------------------------------------------------------------------
// Pack W[K,O] (row-major f32) into WMMA B lane layout, f16:
//   Wp[((tn*KT + kt)*32 + lane)*16 + j] = W[(kt*32 + (lane>>4)*8 + (j&7) + (j>>3)*16)*O + tn*16 + (lane&15)]
// ---------------------------------------------------------------------------
__global__ void k_pack_b(const float* __restrict__ W, _Float16* __restrict__ Wp,
                         int K, int O) {
    int i = blockIdx.x * blockDim.x + threadIdx.x;           // K*O elements
    if (i >= K * O) return;
    const int KT   = K >> 5;
    const int j    = i & 15;
    const int lane = (i >> 4) & 31;
    const int kt   = (i >> 9) % KT;
    const int tn   = i / (KT << 9);
    const int k    = kt * 32 + (lane >> 4) * 8 + (j & 7) + (j >> 3) * 16;
    const int col  = tn * 16 + (lane & 15);
    Wp[i] = (_Float16)W[(long)k * O + col];
}

// ---------------------------------------------------------------------------
// WMMA GEMM: Y[N,O] = X[N,K] @ W[K,O] (+bias)(+relu). K compile-time, K%32==0.
// A layout (ISA 7.12.2): lane l holds row M=l&15; element j<8 -> K=kbA+j,
// j>=8 -> K=kbA+16+(j-8), kbA = kt*32 + (l>>4)*8  => two contiguous 8-float runs.
// B pre-packed (see k_pack_b): one aligned 32B v16h load per k-step.
// C/D: element v -> row = 16*tm + v + 8*(l>>4), col = 16*tn + (l&15).
// Requires N%16==0 (N=100000=6250*16), O%16==0.
// ---------------------------------------------------------------------------
template <int K>
__global__ void k_wmma_gemm(const float* __restrict__ X, const _Float16* __restrict__ Wp,
                            const float* __restrict__ bias, float* __restrict__ Y,
                            int N, int O, int do_relu)
{
    constexpr int KT = K / 32;
    const int wid  = blockIdx.x * (blockDim.x >> 5) + (threadIdx.x >> 5);
    const int lane = threadIdx.x & 31;
    const int colTiles = O >> 4;
    const int tiles = (N >> 4) * colTiles;
    if (wid >= tiles) return;                    // wave-uniform (EXEC stays full)

    const int tm   = wid / colTiles;
    const int tn   = wid - tm * colTiles;
    const int l16  = lane & 15;
    const int half = lane >> 4;
    const int col  = tn * 16 + l16;
    const float* Xrow = X + (long)(tm * 16 + l16) * K;
    const v16h*  Bv   = (const v16h*)Wp + (long)tn * KT * 32 + lane;
    const float  bv   = bias ? bias[col] : 0.f;   // hoisted: one load, no epilogue branches

    v8f acc = {};
    #pragma unroll
    for (int kt = 0; kt < KT; ++kt) {
        const int kbA = kt * 32 + half * 8;
        const float4 a0 = *(const float4*)(Xrow + kbA);
        const float4 a1 = *(const float4*)(Xrow + kbA + 4);
        const float4 a2 = *(const float4*)(Xrow + kbA + 16);
        const float4 a3 = *(const float4*)(Xrow + kbA + 20);
        v16h a;
        a[0]  = (_Float16)a0.x; a[1]  = (_Float16)a0.y; a[2]  = (_Float16)a0.z; a[3]  = (_Float16)a0.w;
        a[4]  = (_Float16)a1.x; a[5]  = (_Float16)a1.y; a[6]  = (_Float16)a1.z; a[7]  = (_Float16)a1.w;
        a[8]  = (_Float16)a2.x; a[9]  = (_Float16)a2.y; a[10] = (_Float16)a2.z; a[11] = (_Float16)a2.w;
        a[12] = (_Float16)a3.x; a[13] = (_Float16)a3.y; a[14] = (_Float16)a3.z; a[15] = (_Float16)a3.w;
        const v16h b = Bv[kt * 32];
        acc = __builtin_amdgcn_wmma_f32_16x16x32_f16(
                  false, a, false, b, (short)0, acc, false, false);
    }

    float* Yout = Y + (long)(tm * 16 + half * 8) * O + col;
    #pragma unroll
    for (int v = 0; v < 8; ++v) {
        float val = acc[v] + bv;
        float rel = fmaxf(val, 0.f);
        Yout[(long)v * O] = do_relu ? rel : val;   // branchless select
    }
}

// ---------------------------------------------------------------------------
// Layer-1 GEMM (K = IN_DIM = 8): trivial VALU kernel, Y[N,64] = X[N,8]@W[8,64]
// ---------------------------------------------------------------------------
__global__ void k_gemm8(const float* __restrict__ X, const float* __restrict__ W,
                        float* __restrict__ Y, int n) {
    int i = blockIdx.x * blockDim.x + threadIdx.x;          // n * 16 chunks of 4 cols
    if (i >= n * 16) return;
    int v = i >> 4, c0 = (i & 15) * 4;
    float4 x0 = *(const float4*)(X + (long)v * 8);
    float4 x1 = *(const float4*)(X + (long)v * 8 + 4);
    float xs[8] = {x0.x, x0.y, x0.z, x0.w, x1.x, x1.y, x1.z, x1.w};
    float4 o = {0.f, 0.f, 0.f, 0.f};
    #pragma unroll
    for (int k = 0; k < 8; ++k) {
        float4 wr = *(const float4*)(W + k * HID + c0);
        o.x += xs[k] * wr.x; o.y += xs[k] * wr.y;
        o.z += xs[k] * wr.z; o.w += xs[k] * wr.w;
    }
    *(float4*)(Y + (long)v * HID + c0) = o;
}

// ---------------------------------------------------------------------------
// Degree / normalization
// ---------------------------------------------------------------------------
__global__ void k_fill1(float* p, int n) {
    int i = blockIdx.x * blockDim.x + threadIdx.x;
    if (i < n) p[i] = 1.0f;                      // self-loop contribution
}
__global__ void k_deg(const int* __restrict__ dst, float* __restrict__ deg, int E) {
    int i = blockIdx.x * blockDim.x + threadIdx.x;
    if (i < E) atomicAdd(&deg[dst[i]], 1.0f);
}
__global__ void k_dinv(const float* __restrict__ deg, float* __restrict__ dinv, int n) {
    int i = blockIdx.x * blockDim.x + threadIdx.x;
    if (i < n) { float d = deg[i]; dinv[i] = d > 0.f ? rsqrtf(d) : 0.f; }
}

// ---------------------------------------------------------------------------
// GCN aggregation (F = 64): agg seeded with self-loop term, then edge atomics,
// then finalize out = res + relu(agg + b).
// ---------------------------------------------------------------------------
__global__ void k_gcn_self(const float* __restrict__ dinv, const float* __restrict__ h,
                           float* __restrict__ agg, int n) {
    int i = blockIdx.x * blockDim.x + threadIdx.x;      // n * 16 chunks of 4
    if (i >= n * 16) return;
    int v = i >> 4, f = (i & 15) * 4;
    float s = dinv[v] * dinv[v];
    float4 hv = *(const float4*)(h + (long)v * HID + f);
    float4 o; o.x = s * hv.x; o.y = s * hv.y; o.z = s * hv.z; o.w = s * hv.w;
    *(float4*)(agg + (long)v * HID + f) = o;
}
__global__ void k_gcn_edge(const int* __restrict__ src, const int* __restrict__ dst,
                           const float* __restrict__ dinv, const float* __restrict__ h,
                           float* __restrict__ agg, int E) {
    int i = blockIdx.x * blockDim.x + threadIdx.x;      // E * 16 chunks of 4
    if (i >= E * 16) return;
    int e = i >> 4, f = (i & 15) * 4;
    int s = src[e], d = dst[e];
    float nm = dinv[s] * dinv[d];
    float4 hv = *(const float4*)(h + (long)s * HID + f);
    float* o = agg + (long)d * HID + f;
    atomicAdd(o + 0, nm * hv.x);
    atomicAdd(o + 1, nm * hv.y);
    atomicAdd(o + 2, nm * hv.z);
    atomicAdd(o + 3, nm * hv.w);
}
__global__ void k_gcn_final(const float* __restrict__ agg, const float* __restrict__ b,
                            const float* __restrict__ res, float* __restrict__ out, int n) {
    int i = blockIdx.x * blockDim.x + threadIdx.x;      // n * 64
    if (i >= n * HID) return;
    float v = fmaxf(agg[i] + b[i & (HID - 1)], 0.f);
    out[i] = (res ? res[i] : 0.f) + v;
}

// ---------------------------------------------------------------------------
// GAT
// ---------------------------------------------------------------------------
__global__ void k_attdot(const float* __restrict__ hg, const float* __restrict__ att_src,
                         const float* __restrict__ att_dst, float* __restrict__ a_src,
                         float* __restrict__ a_dst, int n) {
    int i = blockIdx.x * blockDim.x + threadIdx.x;      // n * HEADS
    if (i >= n * HEADS) return;
    int v = i >> 2, h = i & 3;
    const float* r  = hg + (long)v * GDIM + h * HID;
    const float* as = att_src + h * HID;
    const float* ad = att_dst + h * HID;
    float ss = 0.f, dd = 0.f;
    #pragma unroll 8
    for (int c = 0; c < HID; ++c) { float x = r[c]; ss += x * as[c]; dd += x * ad[c]; }
    a_src[i] = ss; a_dst[i] = dd;
}
__global__ void k_att_minit(const float* __restrict__ a_src, const float* __restrict__ a_dst,
                            float* __restrict__ m, int n) {
    int i = blockIdx.x * blockDim.x + threadIdx.x;      // n * HEADS
    if (i < n * HEADS) m[i] = lrelu02(a_src[i] + a_dst[i]);   // self-loop edge
}
__global__ void k_att_max(const int* __restrict__ src, const int* __restrict__ dst,
                          const float* __restrict__ a_src, const float* __restrict__ a_dst,
                          float* __restrict__ m, int E) {
    int i = blockIdx.x * blockDim.x + threadIdx.x;      // E * HEADS
    if (i >= E * HEADS) return;
    int e = i >> 2, h = i & 3;
    int s = src[e], d = dst[e];
    float ev = lrelu02(a_src[s * HEADS + h] + a_dst[d * HEADS + h]);
    atomicMaxF(&m[d * HEADS + h], ev);
}
__global__ void k_att_self(const float* __restrict__ a_src, const float* __restrict__ a_dst,
                           const float* __restrict__ m, const float* __restrict__ hg,
                           float* __restrict__ denom, float* __restrict__ acc, int n) {
    int i = blockIdx.x * blockDim.x + threadIdx.x;      // n * 64 chunks of 4
    if (i >= n * 64) return;
    int v = i >> 6, ch = i & 63, h = ch >> 4;
    float ex = __expf(lrelu02(a_src[v * HEADS + h] + a_dst[v * HEADS + h]) - m[v * HEADS + h]);
    if ((ch & 15) == 0) denom[v * HEADS + h] = ex;
    float4 hv = *(const float4*)(hg + (long)v * GDIM + ch * 4);
    float4 o; o.x = ex * hv.x; o.y = ex * hv.y; o.z = ex * hv.z; o.w = ex * hv.w;
    *(float4*)(acc + (long)v * GDIM + ch * 4) = o;
}
__global__ void k_att_edge(const int* __restrict__ src, const int* __restrict__ dst,
                           const float* __restrict__ a_src, const float* __restrict__ a_dst,
                           const float* __restrict__ m, const float* __restrict__ hg,
                           float* __restrict__ denom, float* __restrict__ acc, int E) {
    int i = blockIdx.x * blockDim.x + threadIdx.x;      // E * 64 chunks of 4
    if (i >= E * 64) return;
    int e = i >> 6, ch = i & 63, h = ch >> 4;
    int s = src[e], d = dst[e];
    float ex = __expf(lrelu02(a_src[s * HEADS + h] + a_dst[d * HEADS + h]) - m[d * HEADS + h]);
    if ((ch & 15) == 0) atomicAdd(&denom[d * HEADS + h], ex);
    float4 hv = *(const float4*)(hg + (long)s * GDIM + ch * 4);
    float* o = acc + (long)d * GDIM + ch * 4;
    atomicAdd(o + 0, ex * hv.x);
    atomicAdd(o + 1, ex * hv.y);
    atomicAdd(o + 2, ex * hv.z);
    atomicAdd(o + 3, ex * hv.w);
}
__global__ void k_att_final(const float* __restrict__ acc, const float* __restrict__ denom,
                            const float* __restrict__ bg, float* __restrict__ h_att, int n) {
    int i = blockIdx.x * blockDim.x + threadIdx.x;      // n * GDIM
    if (i >= n * GDIM) return;
    int v = i >> 8, col = i & (GDIM - 1), h = col >> 6;
    h_att[i] = acc[i] / (denom[v * HEADS + h] + 1e-16f) + bg[col];
}

// ---------------------------------------------------------------------------
// Classifier tail: logits = z @ wc2 + bc2 ; log_softmax
// ---------------------------------------------------------------------------
__global__ void k_cls_out(const float* __restrict__ z, const float* __restrict__ wc2,
                          const float* __restrict__ bc2, float* __restrict__ out, int n) {
    int v = blockIdx.x * blockDim.x + threadIdx.x;
    if (v >= n) return;
    float l0 = bc2[0], l1 = bc2[1], l2 = bc2[2];
    const float* zr = z + (long)v * HID;
    #pragma unroll 8
    for (int k = 0; k < HID; ++k) {
        float zv = zr[k];
        l0 += zv * wc2[k * OUTD + 0];
        l1 += zv * wc2[k * OUTD + 1];
        l2 += zv * wc2[k * OUTD + 2];
    }
    float mx  = fmaxf(l0, fmaxf(l1, l2));
    float lse = mx + logf(expf(l0 - mx) + expf(l1 - mx) + expf(l2 - mx));
    out[v * OUTD + 0] = l0 - lse;
    out[v * OUTD + 1] = l1 - lse;
    out[v * OUTD + 2] = l2 - lse;
}

// ---------------------------------------------------------------------------
extern "C" void kernel_launch(void* const* d_in, const int* in_sizes, int n_in,
                              void* d_out, int out_size, void* d_ws, size_t ws_size,
                              hipStream_t stream) {
    (void)n_in; (void)out_size; (void)ws_size;

    const float* x    = (const float*)d_in[0];
    const int*   ei   = (const int*)  d_in[1];
    const float* w1   = (const float*)d_in[2];
    const float* b1   = (const float*)d_in[3];
    const float* w2   = (const float*)d_in[4];
    const float* b2   = (const float*)d_in[5];
    const float* w3   = (const float*)d_in[6];
    const float* b3   = (const float*)d_in[7];
    const float* wg   = (const float*)d_in[8];
    const float* bg   = (const float*)d_in[9];
    const float* atts = (const float*)d_in[10];
    const float* attd = (const float*)d_in[11];
    const float* wc1  = (const float*)d_in[12];
    const float* bc1  = (const float*)d_in[13];
    const float* wc2  = (const float*)d_in[14];
    const float* bc2  = (const float*)d_in[15];
    float* out = (float*)d_out;

    const int N = in_sizes[0] / 8;     // IN_DIM = 8
    const int E = in_sizes[1] / 2;
    const int* src = ei;
    const int* dst = ei + E;

    // workspace layout (floats; every segment size is a multiple of 8 floats
    // => all sub-buffers are 32B aligned)
    float* w = (float*)d_ws;
    float* deg   = w;                 w += N;
    float* dinv  = w;                 w += N;
    float* hA    = w;                 w += (long)N * HID;   // current node features
    float* hB    = w;                 w += (long)N * HID;   // GEMM output / z
    float* hg    = w;                 w += (long)N * GDIM;  // GAT features; first N*HID reused as agg
    float* acc   = w;                 w += (long)N * GDIM;  // unnormalized attention sum
    float* a_src = w;                 w += (long)N * HEADS;
    float* a_dst = w;                 w += (long)N * HEADS;
    float* mmax  = w;                 w += (long)N * HEADS;
    float* denom = w;                 w += (long)N * HEADS;
    _Float16* Wp = (_Float16*)w;      // packed B panel (max 256*64 f16 = 32 KB)
    float* agg   = hg;                // GCN aggregation scratch (hg dead until GAT)

    const int gemmWaves = TPB / 32;
    #define GRID(n) cdiv_h((long)(n), TPB), TPB, 0, stream

    // ---- degrees (self-loops included) ----
    k_fill1<<<GRID(N)>>>(deg, N);
    k_deg  <<<GRID(E)>>>(dst, deg, E);
    k_dinv <<<GRID(N)>>>(deg, dinv, N);

    // ---- GCN layer 1 (K=8: plain VALU GEMM; no residual) ----
    {
        k_gemm8    <<<GRID(N * 16)>>>(x, w1, hB, N);
        k_gcn_self <<<GRID(N * 16)>>>(dinv, hB, agg, N);
        k_gcn_edge <<<GRID((long)E * 16)>>>(src, dst, dinv, hB, agg, E);
        k_gcn_final<<<GRID((long)N * HID)>>>(agg, b1, nullptr, hA, N);
    }
    // ---- GCN layers 2,3 (WMMA GEMM K=64; residual) ----
    {
        int tiles = (N >> 4) * (HID >> 4);
        k_pack_b<<<GRID(HID * HID)>>>(w2, Wp, HID, HID);
        k_wmma_gemm<HID><<<cdiv_h(tiles, gemmWaves), TPB, 0, stream>>>(hA, Wp, nullptr, hB, N, HID, 0);
        k_gcn_self <<<GRID(N * 16)>>>(dinv, hB, agg, N);
        k_gcn_edge <<<GRID((long)E * 16)>>>(src, dst, dinv, hB, agg, E);
        k_gcn_final<<<GRID((long)N * HID)>>>(agg, b2, hA, hA, N);

        k_pack_b<<<GRID(HID * HID)>>>(w3, Wp, HID, HID);
        k_wmma_gemm<HID><<<cdiv_h(tiles, gemmWaves), TPB, 0, stream>>>(hA, Wp, nullptr, hB, N, HID, 0);
        k_gcn_self <<<GRID(N * 16)>>>(dinv, hB, agg, N);
        k_gcn_edge <<<GRID((long)E * 16)>>>(src, dst, dinv, hB, agg, E);
        k_gcn_final<<<GRID((long)N * HID)>>>(agg, b3, hA, hA, N);
    }
    // ---- GAT (WMMA GEMM K=64, O=256) ----
    {
        int tiles = (N >> 4) * (GDIM >> 4);
        k_pack_b<<<GRID(HID * GDIM)>>>(wg, Wp, HID, GDIM);
        k_wmma_gemm<HID><<<cdiv_h(tiles, gemmWaves), TPB, 0, stream>>>(hA, Wp, nullptr, hg, N, GDIM, 0);
        k_attdot   <<<GRID(N * HEADS)>>>(hg, atts, attd, a_src, a_dst, N);
        k_att_minit<<<GRID(N * HEADS)>>>(a_src, a_dst, mmax, N);
        k_att_max  <<<GRID((long)E * HEADS)>>>(src, dst, a_src, a_dst, mmax, E);
        k_att_self <<<GRID((long)N * 64)>>>(a_src, a_dst, mmax, hg, denom, acc, N);
        k_att_edge <<<GRID((long)E * 64)>>>(src, dst, a_src, a_dst, mmax, hg, denom, acc, E);
        k_att_final<<<GRID((long)N * GDIM)>>>(acc, denom, bg, hg, N);  // hg := h_att
    }
    // ---- classifier (WMMA GEMM K=256, O=64, fused bias+relu) ----
    {
        int tiles = (N >> 4) * (HID >> 4);
        k_pack_b<<<GRID(GDIM * HID)>>>(wc1, Wp, GDIM, HID);
        k_wmma_gemm<GDIM><<<cdiv_h(tiles, gemmWaves), TPB, 0, stream>>>(hg, Wp, bc1, hB, N, HID, 1);
        k_cls_out  <<<GRID(N)>>>(hB, wc2, bc2, out, N);
    }
    #undef GRID
}
